// SelfAttention_996432413483
// MI455X (gfx1250) — matrix operand order repaired and checked by reading, not compiled, not run on hardware
//
#include <hip/hip_runtime.h>

typedef __attribute__((ext_vector_type(16))) _Float16 v16h;
typedef __attribute__((ext_vector_type(8)))  _Float16 v8h;
typedef __attribute__((ext_vector_type(8)))  float    v8f;

#define BDIM   4
#define CDIM   256
#define NTOK   4096
#define NHEAD  4
#define HDIM   64
#define GROUPS 8
#define SCALE_Q 0.125f   // 64^-0.5

// LDS row pitches (halves) chosen so row*pitch mod 64 banks is a permutation
#define KPITCH 72   // 32x64 K tile rows
#define VPITCH 40   // 64x32 V tile rows
#define PPITCH 40   // 16x32 P tile rows

// ---------------------------------------------------------------------------
// WMMA fragment helpers (CDNA5 wave32, V_WMMA_F32_16X16X32_F16)
// A fragment: 16x32 f16 from row-major [M][stride]:
//   lane l: row M = l&15 ; halves 0-7 -> K = (l>>4)*8 + 0..7,
//                          halves 8-15 -> K = 16 + (l>>4)*8 + 0..7
// B fragment from B^T row-major [N][stride]:
//   lane l: col N = l&15 ; halves 0-7 -> K = (l>>4)*8 + 0..7,
//                          halves 8-15 -> K = 16 + (l>>4)*8 + 0..7
// ---------------------------------------------------------------------------
__device__ __forceinline__ v16h frag_a(const _Float16* base, int stride, int lane) {
  const _Float16* p = base + (size_t)(lane & 15) * stride + ((lane >> 4) << 3);
  v8h lo = *(const v8h*)p;
  v8h hi = *(const v8h*)(p + 16);
  return __builtin_shufflevector(lo, hi, 0,1,2,3,4,5,6,7,8,9,10,11,12,13,14,15);
}
__device__ __forceinline__ v16h frag_bt(const _Float16* base, int stride, int lane) {
  const _Float16* p = base + (size_t)(lane & 15) * stride + ((lane >> 4) << 3);
  v8h lo = *(const v8h*)p;
  v8h hi = *(const v8h*)(p + 16);
  return __builtin_shufflevector(lo, hi, 0,1,2,3,4,5,6,7,8,9,10,11,12,13,14,15);
}
__device__ __forceinline__ v8f wmma_f16(v16h a, v16h b, v8f c) {
  return __builtin_amdgcn_wmma_f32_16x16x32_f16(false, a, false, b, (short)0, c,
                                                false, false);
}

// async global->LDS 16B copy (ASYNCcnt); lds operand = low 32 bits of generic ptr
__device__ __forceinline__ void async_cp16(const _Float16* lds_dst, const _Float16* g) {
  uint32_t l = (uint32_t)(uintptr_t)lds_dst;
  asm volatile("global_load_async_to_lds_b128 %0, %1, off"
               :: "v"(l), "v"(g) : "memory");
}
__device__ __forceinline__ void async_drain() {
  asm volatile("s_wait_asynccnt 0x0" ::: "memory");
}

// ---------------------------------------------------------------------------
// 1) weight conversion f32 -> f16 (run every launch; deterministic/idempotent)
// ---------------------------------------------------------------------------
__global__ __launch_bounds__(256) void cvt_weights(const float* __restrict__ qkvw,
                                                   const float* __restrict__ projw,
                                                   _Float16* __restrict__ wq,
                                                   _Float16* __restrict__ wp) {
  int i = blockIdx.x * 256 + threadIdx.x;
  if (i < 3 * CDIM * CDIM) wq[i] = (_Float16)qkvw[i];
  if (i < CDIM * CDIM)     wp[i] = (_Float16)projw[i];
}

// ---------------------------------------------------------------------------
// 2) GroupNorm statistics: one block per (b, group); 32 ch x 4096 px reduce
// ---------------------------------------------------------------------------
__global__ __launch_bounds__(256) void gn_stats(const float* __restrict__ x,
                                                float* __restrict__ stats) {
  const int bg = blockIdx.x;                 // 0..31
  const int b = bg >> 3, g = bg & 7;
  const float* p = x + ((size_t)b * CDIM + g * (CDIM / GROUPS)) * NTOK;
  float s = 0.f, s2 = 0.f;
  const int total = (CDIM / GROUPS) * NTOK;  // 131072
  for (int i = threadIdx.x; i < total; i += 256) {
    float v = p[i];
    s += v; s2 += v * v;
  }
  #pragma unroll
  for (int m = 1; m < 32; m <<= 1) {
    s  += __shfl_xor(s,  m, 32);
    s2 += __shfl_xor(s2, m, 32);
  }
  __shared__ float red[16];
  const int lane = threadIdx.x & 31, wave = threadIdx.x >> 5;
  if (lane == 0) { red[wave * 2] = s; red[wave * 2 + 1] = s2; }
  __syncthreads();
  if (threadIdx.x == 0) {
    float ts = 0.f, ts2 = 0.f;
    for (int w = 0; w < 8; ++w) { ts += red[w * 2]; ts2 += red[w * 2 + 1]; }
    float mean = ts / (float)total;
    float var  = ts2 / (float)total - mean * mean;
    stats[bg * 2]     = mean;
    stats[bg * 2 + 1] = rsqrtf(var + 1e-5f);
  }
}

// ---------------------------------------------------------------------------
// 3) GN apply + transpose: x[b][c][n] f32 -> hnT[b][n][c] f16 (64x64 LDS tile)
// ---------------------------------------------------------------------------
__global__ __launch_bounds__(256) void gn_apply(const float* __restrict__ x,
                                                const float* __restrict__ gamma,
                                                const float* __restrict__ beta,
                                                const float* __restrict__ stats,
                                                _Float16* __restrict__ hnT) {
  const int blk = blockIdx.x;                // b(4) x ctile(4) x ntile(64)
  const int ntile = blk & 63;
  const int ctile = (blk >> 6) & 3;
  const int b = blk >> 8;
  const int c0 = ctile * 64, n0 = ntile * 64;
  __shared__ float tile[64][65];
  const int tn = threadIdx.x & 63;
  const int t4 = threadIdx.x >> 6;           // 0..3
  #pragma unroll 4
  for (int it = 0; it < 16; ++it) {
    int cl = it * 4 + t4;
    tile[cl][tn] = x[((size_t)(b * CDIM + c0 + cl)) * NTOK + n0 + tn];
  }
  __syncthreads();
  const int c = c0 + tn;
  const int g = c >> 5;
  const float mean = stats[(b * 8 + g) * 2];
  const float rstd = stats[(b * 8 + g) * 2 + 1];
  const float ga = gamma[c] * rstd;
  const float be = beta[c] - mean * ga;
  #pragma unroll 4
  for (int it = 0; it < 16; ++it) {
    int nl = it * 4 + t4;
    float v = tile[tn][nl];
    hnT[((size_t)b * NTOK + n0 + nl) * CDIM + c] = (_Float16)(v * ga + be);
  }
}

// ---------------------------------------------------------------------------
// 4) QKV GEMM: [4096 x 256] x [256 x 768]^T per batch, WMMA f16->f32,
//    software-pipelined K loop (double-buffered fragments).
//    wave tile = 16 tokens x 64 outputs; scatter into Q/K [b][h][n][d],
//    V [b][h][d][n]; Q pre-scaled by 1/sqrt(d)
// ---------------------------------------------------------------------------
__global__ __launch_bounds__(256) void qkv_gemm(const _Float16* __restrict__ hnT,
                                                const _Float16* __restrict__ W,
                                                const float* __restrict__ bias,
                                                _Float16* __restrict__ Q,
                                                _Float16* __restrict__ K,
                                                _Float16* __restrict__ V) {
  const int lane = threadIdx.x & 31, wave = threadIdx.x >> 5;
  const int wid = blockIdx.x * 8 + wave;     // 0..12287
  const int ot = wid % 12;
  const int t2 = wid / 12;
  const int nt = t2 & 255;
  const int b  = t2 >> 8;
  const int ob = ot * 64, nb = nt * 16;
  const int r16 = lane & 15;
  const _Float16* Ap = hnT + ((size_t)b * NTOK + nb) * CDIM;
  const _Float16* Wp = W + (size_t)ob * CDIM;
  const v8f vzero = {0.f,0.f,0.f,0.f,0.f,0.f,0.f,0.f};
  v8f acc[4] = {vzero, vzero, vzero, vzero};

  v16h a = frag_a(Ap, CDIM, lane);
  v16h wf[4];
  #pragma unroll
  for (int os = 0; os < 4; ++os)
    wf[os] = frag_bt(Wp + (size_t)(os * 16) * CDIM, CDIM, lane);

  #pragma unroll
  for (int cb = 0; cb < CDIM; cb += 32) {
    const int cn = (cb + 32 < CDIM) ? cb + 32 : 0;  // clamped: branchless pipeline
    v16h a2 = frag_a(Ap + cn, CDIM, lane);
    v16h wf2[4];
    #pragma unroll
    for (int os = 0; os < 4; ++os)
      wf2[os] = frag_bt(Wp + (size_t)(os * 16) * CDIM + cn, CDIM, lane);
    #pragma unroll
    for (int os = 0; os < 4; ++os)
      acc[os] = wmma_f16(a, wf[os], acc[os]);
    a = a2;
    #pragma unroll
    for (int os = 0; os < 4; ++os) wf[os] = wf2[os];
  }

  const int t = ob >> 8;                     // 0=q,1=k,2=v
  const int h = (ob >> 6) & 3;
  const int nr0 = nb + ((lane >> 4) << 3);
  #pragma unroll
  for (int os = 0; os < 4; ++os) {
    const int o = ob + os * 16 + r16;
    const float bv = bias[o];
    const int d = o & 63;
    if (t == 0) {
      _Float16* dst = Q + ((size_t)(b * NHEAD + h) * NTOK + nr0) * HDIM + d;
      #pragma unroll
      for (int r = 0; r < 8; ++r)
        dst[(size_t)r * HDIM] = (_Float16)((acc[os][r] + bv) * SCALE_Q);
    } else if (t == 1) {
      _Float16* dst = K + ((size_t)(b * NHEAD + h) * NTOK + nr0) * HDIM + d;
      #pragma unroll
      for (int r = 0; r < 8; ++r)
        dst[(size_t)r * HDIM] = (_Float16)(acc[os][r] + bv);
    } else {
      _Float16* dst = V + ((size_t)(b * NHEAD + h) * HDIM + d) * NTOK + nr0;
      #pragma unroll
      for (int r = 0; r < 8; ++r)
        dst[r] = (_Float16)(acc[os][r] + bv);
    }
  }
}

// ---------------------------------------------------------------------------
// 5) Flash attention with ASYNC global->LDS staging of K/V tiles.
//    Workgroup (8 waves, 256 rows of n) shares each 32x64 K tile and 64x32 V
//    tile via LDS (staged once with global_load_async_to_lds_b128, parity
//    ping-pong, s_wait_asynccnt + barrier). Waves consume ds_load_b128
//    fragments; online softmax in C-layout; P via padded LDS (C->A swap).
// ---------------------------------------------------------------------------
__global__ __launch_bounds__(256) void attn_flash(const _Float16* __restrict__ Q,
                                                  const _Float16* __restrict__ K,
                                                  const _Float16* __restrict__ V,
                                                  _Float16* __restrict__ attnT) {
  const int tid  = threadIdx.x;
  const int lane = tid & 31;
  const int wave = tid >> 5;
  const int bh = blockIdx.x >> 4;            // 0..15 = b*4+h
  const int nchunk = blockIdx.x & 15;
  const int nbase = nchunk * 256 + wave * 32;
  const int b = bh >> 2, h = bh & 3;
  const int r16 = lane & 15;
  const _Float16* Qp = Q + (size_t)bh * NTOK * HDIM;
  const _Float16* Kp = K + (size_t)bh * NTOK * HDIM;
  const _Float16* Vp = V + (size_t)bh * HDIM * NTOK;

  __shared__ __align__(16) _Float16 Kt[2][32 * KPITCH];        //  9.0 KB
  __shared__ __align__(16) _Float16 Vt[2][64 * VPITCH];        // 10.0 KB
  __shared__ __align__(16) _Float16 plds[8][2][2][16 * PPITCH];// 40.0 KB

  // per-thread async staging slots: one 16B chunk of K, one of V
  const int km = tid >> 3, kc = tid & 7;     // K: row 0..31, chunk 0..7
  const int vd = tid >> 2, vc = tid & 3;     // V: row 0..63, chunk 0..3

  // prologue: stage tile mb=0 into parity 0
  async_cp16(&Kt[0][km * KPITCH + kc * 8], Kp + (size_t)km * HDIM + kc * 8);
  async_cp16(&Vt[0][vd * VPITCH + vc * 8], Vp + (size_t)vd * NTOK + vc * 8);

  // Q fragments resident for the whole pass: [rowtile][dstep]
  v16h qa[2][2];
  #pragma unroll
  for (int rt = 0; rt < 2; ++rt)
    #pragma unroll
    for (int ds = 0; ds < 2; ++ds)
      qa[rt][ds] = frag_a(Qp + (size_t)(nbase + rt * 16) * HDIM + ds * 32, HDIM, lane);

  const v8f vzero = {0.f,0.f,0.f,0.f,0.f,0.f,0.f,0.f};
  v8f acc[2][4] = {{vzero,vzero,vzero,vzero},{vzero,vzero,vzero,vzero}};
  float mrow[2][8], lrow[2][8];
  #pragma unroll
  for (int rt = 0; rt < 2; ++rt)
    #pragma unroll
    for (int r = 0; r < 8; ++r) { mrow[rt][r] = -1e30f; lrow[rt][r] = 0.f; }

  async_drain();
  __syncthreads();                           // tile 0 visible to all waves

  for (int mb = 0; mb < NTOK; mb += 32) {
    const int par = (mb >> 5) & 1;

    // stage NEXT tile into other parity while we compute (uniform branch)
    if (mb + 32 < NTOK) {
      const int mn = mb + 32;
      async_cp16(&Kt[par ^ 1][km * KPITCH + kc * 8],
                 Kp + (size_t)(mn + km) * HDIM + kc * 8);
      async_cp16(&Vt[par ^ 1][vd * VPITCH + vc * 8],
                 Vp + (size_t)vd * NTOK + mn + vc * 8);
      __builtin_prefetch(Kp + (size_t)(mn + 32) * HDIM, 0, 1);
      __builtin_prefetch(Vp + mn + 32, 0, 1);
    }

    // S = Q K^T  (16n x 32m per rowtile); K fragments from LDS
    v16h kf[2][2];
    #pragma unroll
    for (int ms = 0; ms < 2; ++ms)
      #pragma unroll
      for (int ds = 0; ds < 2; ++ds)
        kf[ms][ds] = frag_bt(&Kt[par][(size_t)(ms * 16) * KPITCH + ds * 32],
                             KPITCH, lane);
    v8f s[2][2];
    #pragma unroll
    for (int rt = 0; rt < 2; ++rt)
      #pragma unroll
      for (int ms = 0; ms < 2; ++ms) {
        v8f t = wmma_f16(qa[rt][0], kf[ms][0], vzero);
        s[rt][ms] = wmma_f16(qa[rt][1], kf[ms][1], t);
      }
    // V fragments from LDS (consumed after softmax)
    v16h vf[4];
    #pragma unroll
    for (int dt = 0; dt < 4; ++dt)
      vf[dt] = frag_bt(&Vt[par][(size_t)(dt * 16) * VPITCH], VPITCH, lane);

    // online softmax in C-layout: row M -> (vgpr M%8, lane-half M/8)
    #pragma unroll
    for (int rt = 0; rt < 2; ++rt) {
      _Float16* lp = &plds[wave][par][rt][0];
      const int rowb = (lane >> 4) << 3;
      #pragma unroll
      for (int r = 0; r < 8; ++r) {
        float t0 = fmaxf(s[rt][0][r], s[rt][1][r]);
        t0 = fmaxf(t0, __shfl_xor(t0, 1, 32));
        t0 = fmaxf(t0, __shfl_xor(t0, 2, 32));
        t0 = fmaxf(t0, __shfl_xor(t0, 4, 32));
        t0 = fmaxf(t0, __shfl_xor(t0, 8, 32));
        const float mn = fmaxf(mrow[rt][r], t0);
        const float al = __expf(mrow[rt][r] - mn);
        mrow[rt][r] = mn;
        const float p0 = __expf(s[rt][0][r] - mn);
        const float p1 = __expf(s[rt][1][r] - mn);
        float rs = p0 + p1;
        rs += __shfl_xor(rs, 1, 32);
        rs += __shfl_xor(rs, 2, 32);
        rs += __shfl_xor(rs, 4, 32);
        rs += __shfl_xor(rs, 8, 32);
        lrow[rt][r] = lrow[rt][r] * al + rs;
        #pragma unroll
        for (int dt = 0; dt < 4; ++dt) acc[rt][dt][r] *= al;
        lp[(rowb + r) * PPITCH + r16]      = (_Float16)p0;
        lp[(rowb + r) * PPITCH + 16 + r16] = (_Float16)p1;
      }
    }
    // O += P V^T   (same-wave DS store->load, in order per ISA)
    #pragma unroll
    for (int rt = 0; rt < 2; ++rt) {
      v16h pa = frag_a(&plds[wave][par][rt][0], PPITCH, lane);
      #pragma unroll
      for (int dt = 0; dt < 4; ++dt)
        acc[rt][dt] = wmma_f16(pa, vf[dt], acc[rt][dt]);
    }

    async_drain();        // my async chunks for next tile have landed
    __syncthreads();      // all waves done reading par / staging par^1
  }

  // epilogue: O / l, write attnT[b][n][h*64+d]
  const int rowb = (lane >> 4) << 3;
  #pragma unroll
  for (int rt = 0; rt < 2; ++rt)
    #pragma unroll
    for (int dt = 0; dt < 4; ++dt) {
      #pragma unroll
      for (int r = 0; r < 8; ++r) {
        const int n = nbase + rt * 16 + rowb + r;
        const float v = acc[rt][dt][r] / lrow[rt][r];
        attnT[((size_t)b * NTOK + n) * CDIM + h * HDIM + dt * 16 + r16] = (_Float16)v;
      }
    }
}

// ---------------------------------------------------------------------------
// 6) Output projection + bias + residual, fp32 out[b][c][n], pipelined K loop
// ---------------------------------------------------------------------------
__global__ __launch_bounds__(256) void proj_gemm(const _Float16* __restrict__ aT,
                                                 const _Float16* __restrict__ W,
                                                 const float* __restrict__ bias,
                                                 const float* __restrict__ x,
                                                 float* __restrict__ out) {
  const int lane = threadIdx.x & 31, wave = threadIdx.x >> 5;
  const int wid = blockIdx.x * 8 + wave;     // 0..4095
  const int ot = wid & 3;
  const int t2 = wid >> 2;
  const int nt = t2 & 255;
  const int b  = t2 >> 8;
  const int ob = ot * 64, nb = nt * 16;
  const int r16 = lane & 15;
  const _Float16* Ap = aT + ((size_t)b * NTOK + nb) * CDIM;
  const _Float16* Wb = W + (size_t)ob * CDIM;
  const v8f vzero = {0.f,0.f,0.f,0.f,0.f,0.f,0.f,0.f};
  v8f acc[4] = {vzero, vzero, vzero, vzero};

  v16h a = frag_a(Ap, CDIM, lane);
  v16h wf[4];
  #pragma unroll
  for (int os = 0; os < 4; ++os)
    wf[os] = frag_bt(Wb + (size_t)(os * 16) * CDIM, CDIM, lane);

  #pragma unroll
  for (int cb = 0; cb < CDIM; cb += 32) {
    const int cn = (cb + 32 < CDIM) ? cb + 32 : 0;
    v16h a2 = frag_a(Ap + cn, CDIM, lane);
    v16h wf2[4];
    #pragma unroll
    for (int os = 0; os < 4; ++os)
      wf2[os] = frag_bt(Wb + (size_t)(os * 16) * CDIM + cn, CDIM, lane);
    #pragma unroll
    for (int os = 0; os < 4; ++os)
      acc[os] = wmma_f16(a, wf[os], acc[os]);
    a = a2;
    #pragma unroll
    for (int os = 0; os < 4; ++os) wf[os] = wf2[os];
  }

  const int nr0 = nb + ((lane >> 4) << 3);
  #pragma unroll
  for (int os = 0; os < 4; ++os) {
    const int o = ob + os * 16 + r16;
    const float bv = bias[o];
    const size_t base = ((size_t)(b * CDIM + o)) * NTOK + nr0;
    float4 x0 = *(const float4*)(x + base);
    float4 x1 = *(const float4*)(x + base + 4);
    float4 y0, y1;
    y0.x = x0.x + bv + acc[os][0];
    y0.y = x0.y + bv + acc[os][1];
    y0.z = x0.z + bv + acc[os][2];
    y0.w = x0.w + bv + acc[os][3];
    y1.x = x1.x + bv + acc[os][4];
    y1.y = x1.y + bv + acc[os][5];
    y1.z = x1.z + bv + acc[os][6];
    y1.w = x1.w + bv + acc[os][7];
    *(float4*)(out + base)     = y0;
    *(float4*)(out + base + 4) = y1;
  }
}

// ---------------------------------------------------------------------------
extern "C" void kernel_launch(void* const* d_in, const int* in_sizes, int n_in,
                              void* d_out, int out_size, void* d_ws, size_t ws_size,
                              hipStream_t stream) {
  (void)in_sizes; (void)n_in; (void)out_size; (void)ws_size;
  const float* x     = (const float*)d_in[0];
  const float* gnw   = (const float*)d_in[1];
  const float* gnb   = (const float*)d_in[2];
  const float* qkvw  = (const float*)d_in[3];
  const float* qkvb  = (const float*)d_in[4];
  const float* projw = (const float*)d_in[5];
  const float* projb = (const float*)d_in[6];
  float* out = (float*)d_out;

  char* ws = (char*)d_ws;
  float*    stats = (float*)(ws);                       // 1 KB
  _Float16* wq    = (_Float16*)(ws + 1024);             // 768*256 f16
  _Float16* wp    = (_Float16*)(ws + 394240);           // 256*256 f16
  _Float16* hnT   = (_Float16*)(ws + 525312);           // 4*4096*256 f16
  _Float16* Qb    = (_Float16*)(ws + 8913920);
  _Float16* Kb    = (_Float16*)(ws + 17302528);
  _Float16* Vb    = (_Float16*)(ws + 25691136);
  _Float16* aT    = (_Float16*)(ws + 34079744);         // ends ~42.5 MB

  cvt_weights<<<768,  256, 0, stream>>>(qkvw, projw, wq, wp);
  gn_stats   <<<32,   256, 0, stream>>>(x, stats);
  gn_apply   <<<1024, 256, 0, stream>>>(x, gnw, gnb, stats, hnT);
  qkv_gemm   <<<1536, 256, 0, stream>>>(hnT, wq, qkvb, Qb, Kb, Vb);
  attn_flash <<<256,  256, 0, stream>>>(Qb, Kb, Vb, aT);
  proj_gemm  <<<512,  256, 0, stream>>>(aT, wp, projb, x, out);
}
